// CategorySpecificLinear_9809705304280
// MI455X (gfx1250) — matrix operand order repaired and checked by reading, not compiled, not run on hardware
//
#include <hip/hip_runtime.h>

// CategorySpecificLinear for MI455X (gfx1250, wave32).
// out[b,s,h] = sum_i x[b,s,i] * W[cat[b],i,h] + bias[cat[b],h]
// B=64, SEQ=256, IN=1024, HID=4096.
// f32 in/out, computed with split-precision bf16x3 on V_WMMA_F32_16X16X32_BF16.
// Double-buffered LDS pipeline: chunk i+1 global loads fly while chunk i
// WMMAs run; one barrier per K-chunk.

typedef __attribute__((ext_vector_type(8)))  float  v8f;
typedef __attribute__((ext_vector_type(8)))  __bf16 v8bf;
typedef __attribute__((ext_vector_type(16))) __bf16 v16bf;

#define BATCH     64
#define SEQLEN    256
#define IN_DIM    1024
#define HID_DIM   4096
#define KC        32          // K chunk per pipeline stage (one WMMA k-depth)
#define NCHUNK    (IN_DIM / KC)
#define XS_STRIDE 40          // 32 + 8 bf16 pad: 80B rows, 16B aligned, bank-safe
#define WS_STRIDE 40
#define XPLANE    (64  * XS_STRIDE)   // 2560 bf16
#define WPLANE    (128 * WS_STRIDE)   // 5120 bf16

__global__ __launch_bounds__(256)
void cat_linear_wmma_bf16x3_db(const float* __restrict__ x,
                               const int*   __restrict__ cat_ids,
                               const float* __restrict__ W,
                               const float* __restrict__ bias,
                               float*       __restrict__ out)
{
    // Double-buffered hi/lo split planes. X: row-major (M x K).
    // W: transposed (N x K) so B fragments are contiguous-K b128 loads.
    __shared__ __bf16 Xs_hi[2][XPLANE];   // 10240 B
    __shared__ __bf16 Xs_lo[2][XPLANE];   // 10240 B
    __shared__ __bf16 Ws_hi[2][WPLANE];   // 20480 B
    __shared__ __bf16 Ws_lo[2][WPLANE];   // 20480 B  (total 61440 B)

    const int tid    = threadIdx.x;
    const int batch  = blockIdx.z;
    const int blockM = blockIdx.y * 64;    // SEQ tile
    const int blockN = blockIdx.x * 128;   // HID tile

    const int cat = cat_ids[batch];

    const float* Xb = x    + ((size_t)batch * SEQLEN + blockM) * IN_DIM;
    const float* Wc = W    + (size_t)cat * IN_DIM * HID_DIM + blockN;
    const float* bb = bias + (size_t)cat * HID_DIM;
    float*     outb = out  + ((size_t)batch * SEQLEN + blockM) * (size_t)HID_DIM + blockN;

    // 8 wave32s -> 2 (M) x 4 (N); each wave computes a 32x32 output tile.
    const int wave = tid >> 5;
    const int lane = tid & 31;
    const int half = lane >> 4;
    const int lrow = lane & 15;
    const int wm   = wave >> 2;   // 0..1
    const int wn   = wave & 3;    // 0..3

    // X staging: 4 threads/row, 8 consecutive k each (-> b128-mergeable).
    const int xrow = tid >> 2;           // 0..63
    const int xcol = (tid & 3) * 8;      // 0,8,16,24
    // W staging: thread owns one n-column and a 16-deep k run; per-k reads are
    // 128-lane coalesced b32 bursts.
    const int wcol = tid & 127;
    const int kh   = tid >> 7;           // 0..1 -> k run [kh*16, kh*16+16)

    float xreg[8];
    float wreg[16];

    auto load_chunk = [&](int k0) {
        #pragma unroll
        for (int e = 0; e < 8; ++e)
            xreg[e] = Xb[(size_t)xrow * IN_DIM + k0 + xcol + e];
        #pragma unroll
        for (int e = 0; e < 16; ++e)
            wreg[e] = Wc[(size_t)(k0 + kh * 16 + e) * HID_DIM + wcol];
    };

    auto stage_chunk = [&](int buf) {
        // X: 8 elements -> one v8bf per plane.
        v8bf xhi, xlo;
        #pragma unroll
        for (int e = 0; e < 8; ++e) {
            const float v  = xreg[e];
            const __bf16 h = (__bf16)v;
            xhi[e] = h;
            xlo[e] = (__bf16)(v - (float)h);
        }
        *(v8bf*)&Xs_hi[buf][xrow * XS_STRIDE + xcol] = xhi;
        *(v8bf*)&Xs_lo[buf][xrow * XS_STRIDE + xcol] = xlo;
        // W: 16 elements -> two v8bf per plane (transposed, k-contiguous).
        #pragma unroll
        for (int j = 0; j < 2; ++j) {
            v8bf whi, wlo;
            #pragma unroll
            for (int e = 0; e < 8; ++e) {
                const float v  = wreg[j * 8 + e];
                const __bf16 h = (__bf16)v;
                whi[e] = h;
                wlo[e] = (__bf16)(v - (float)h);
            }
            *(v8bf*)&Ws_hi[buf][wcol * WS_STRIDE + kh * 16 + j * 8] = whi;
            *(v8bf*)&Ws_lo[buf][wcol * WS_STRIDE + kh * 16 + j * 8] = wlo;
        }
    };

    v8f acc[2][2];
    #pragma unroll
    for (int i = 0; i < 2; ++i)
        #pragma unroll
        for (int j = 0; j < 2; ++j)
            acc[i][j] = (v8f){0.f,0.f,0.f,0.f,0.f,0.f,0.f,0.f};

    // Prologue: fill buffer 0.
    load_chunk(0);
    stage_chunk(0);
    __syncthreads();

    for (int i = 0; i < NCHUNK; ++i) {
        const int  buf  = i & 1;
        const bool more = (i + 1) < NCHUNK;   // block-uniform

        // Kick off next chunk's global loads (in flight during WMMAs).
        if (more) load_chunk((i + 1) * KC);

        // A fragments: lane holds contiguous K in [16h, 16h+16).
        v16bf ahi[2], alo[2];
        #pragma unroll
        for (int mt = 0; mt < 2; ++mt) {
            const int row = wm * 32 + mt * 16 + lrow;
            const __bf16* ph = &Xs_hi[buf][row * XS_STRIDE];
            const __bf16* pl = &Xs_lo[buf][row * XS_STRIDE];
            v8bf h0 = *(const v8bf*)(ph + 16 * half);
            v8bf h1 = *(const v8bf*)(ph + 16 * half + 8);
            v8bf l0 = *(const v8bf*)(pl + 16 * half);
            v8bf l1 = *(const v8bf*)(pl + 16 * half + 8);
            ahi[mt] = __builtin_shufflevector(h0, h1, 0,1,2,3,4,5,6,7,8,9,10,11,12,13,14,15);
            alo[mt] = __builtin_shufflevector(l0, l1, 0,1,2,3,4,5,6,7,8,9,10,11,12,13,14,15);
        }

        // B fragments: lane holds contiguous K in [16h, 16h+16) for column N.
        v16bf bhi[2], blo[2];
        #pragma unroll
        for (int nt = 0; nt < 2; ++nt) {
            const int col = wn * 32 + nt * 16 + lrow;
            const __bf16* ph = &Ws_hi[buf][col * WS_STRIDE + 16 * half];
            const __bf16* pl = &Ws_lo[buf][col * WS_STRIDE + 16 * half];
            v8bf h0 = *(const v8bf*)(ph);
            v8bf h1 = *(const v8bf*)(ph + 8);
            v8bf l0 = *(const v8bf*)(pl);
            v8bf l1 = *(const v8bf*)(pl + 8);
            bhi[nt] = __builtin_shufflevector(h0, h1, 0,1,2,3,4,5,6,7,8,9,10,11,12,13,14,15);
            blo[nt] = __builtin_shufflevector(l0, l1, 0,1,2,3,4,5,6,7,8,9,10,11,12,13,14,15);
        }

        // bf16x3: hi*hi + hi*lo + lo*hi (lo*lo dropped, ~2^-16 relative).
        #pragma unroll
        for (int mt = 0; mt < 2; ++mt)
            #pragma unroll
            for (int nt = 0; nt < 2; ++nt) {
                acc[mt][nt] = __builtin_amdgcn_wmma_f32_16x16x32_bf16(
                    false, ahi[mt], false, bhi[nt], (short)0, acc[mt][nt], false, false);
                acc[mt][nt] = __builtin_amdgcn_wmma_f32_16x16x32_bf16(
                    false, ahi[mt], false, blo[nt], (short)0, acc[mt][nt], false, false);
                acc[mt][nt] = __builtin_amdgcn_wmma_f32_16x16x32_bf16(
                    false, alo[mt], false, bhi[nt], (short)0, acc[mt][nt], false, false);
            }

        // Convert + store next chunk into the other buffer, then one barrier.
        if (more) stage_chunk(buf ^ 1);
        __syncthreads();
    }

    // Epilogue: bias add + store. C/D layout: VGPR r -> row r + 8*half, col = lrow.
    #pragma unroll
    for (int nt = 0; nt < 2; ++nt) {
        const int col = wn * 32 + nt * 16 + lrow;
        const float bv = bb[blockN + col];
        #pragma unroll
        for (int mt = 0; mt < 2; ++mt) {
            #pragma unroll
            for (int r = 0; r < 8; ++r) {
                const int row = wm * 32 + mt * 16 + half * 8 + r;
                outb[(size_t)row * HID_DIM + col] = acc[mt][nt][r] + bv;
            }
        }
    }
}

extern "C" void kernel_launch(void* const* d_in, const int* in_sizes, int n_in,
                              void* d_out, int out_size, void* d_ws, size_t ws_size,
                              hipStream_t stream) {
    (void)in_sizes; (void)n_in; (void)out_size; (void)d_ws; (void)ws_size;
    const float* x   = (const float*)d_in[0];
    const int*   cid = (const int*)d_in[1];   // integer input per harness contract
    const float* W   = (const float*)d_in[2];
    const float* b   = (const float*)d_in[3];
    float*       out = (float*)d_out;

    dim3 grid(HID_DIM / 128, SEQLEN / 64, BATCH);  // (32, 4, 64)
    cat_linear_wmma_bf16x3_db<<<grid, 256, 0, stream>>>(x, cid, W, b, out);
}
